// NodeGNN_63084479644011
// MI455X (gfx1250) — matrix operand order repaired and checked by reading, not compiled, not run on hardware
//
#include <hip/hip_runtime.h>

typedef __attribute__((ext_vector_type(16))) __bf16    v16bf;
typedef __attribute__((ext_vector_type(8)))  float     v8f;
typedef __attribute__((ext_vector_type(8)))  unsigned  v8u;
typedef __attribute__((ext_vector_type(4)))  unsigned  v4u;

#define KDIM 128      // inner dim of every GEMM in this net
#define KC4  4        // KDIM / 32

// ---------------------------------------------------------------------------
// Pack a weight matrix into the WMMA bf16 B-fragment layout, split hi/lo.
//   dst layout (dwords): [n_tile][k_chunk][sel(hi=0,lo=1)][lane 0..31][dword 0..7]
//   B-matrix (32x16 bf16) register layout: lane L -> n = L%16, g = L/16,
//   dword j half h -> k = (j/4)*16 + g*8 + (j%4)*2 + h        (per CDNA5 ISA B layout)
// transposed=1: W is [OUT][K] (torch Linear weight, used as x @ W^T)
// transposed=0: W is [K][OUT]
// ---------------------------------------------------------------------------
__global__ void pack_weight_kernel(const float* __restrict__ W,
                                   unsigned* __restrict__ dst,
                                   int OUT, int transposed)
{
    int idx = blockIdx.x * 256 + threadIdx.x;
    int total = (OUT / 16) * KC4 * 32 * 8;
    if (idx >= total) return;
    int v    = idx & 7;
    int lane = (idx >> 3) & 31;
    int kc   = (idx >> 8) % KC4;
    int nt   = (idx >> 8) / KC4;
    int g = lane >> 4, n = lane & 15;
    int o = nt * 16 + n;
    unsigned hi = 0, lo = 0;
#pragma unroll
    for (int h = 0; h < 2; ++h) {
        int k = kc * 32 + (v >> 2) * 16 + g * 8 + (v & 3) * 2 + h;
        float f = transposed ? W[(size_t)o * KDIM + k] : W[(size_t)k * OUT + o];
        __bf16 bh = (__bf16)f;
        __bf16 bl = (__bf16)(f - (float)bh);
        hi |= (unsigned)__builtin_bit_cast(unsigned short, bh) << (16 * h);
        lo |= (unsigned)__builtin_bit_cast(unsigned short, bl) << (16 * h);
    }
    size_t base = ((((size_t)nt * KC4 + kc) * 2) * 32 + lane) * 8 + v;
    dst[base]       = hi;
    dst[base + 256] = lo;    // sel=1 block (+32 lanes * 8 dwords)
}

// ---------------------------------------------------------------------------
// GEMM: Y[128 rows x OUT] per block = (X (+relu_in)) @ Wpacked (+bias).
// 8 waves; wave w owns M-tile w (16 rows) and loops over all OUT/16 N-tiles.
// fp32 -> bf16 hi/lo split is done ONCE, cooperatively, into LDS in
// A-fragment dword order, so the compute loop is pure ds_load + v_wmma.
// ---------------------------------------------------------------------------
template <int OUT>
__global__ void gemm_wmma_kernel(const float* __restrict__ X,
                                 const unsigned* __restrict__ Wp,
                                 const float* __restrict__ bias,
                                 float* __restrict__ Y,
                                 int N, int add_bias, int relu_in)
{
    const int RS = 68;                       // row stride in dwords (64 + 4 pad)
    __shared__ unsigned Ahi[128 * RS];
    __shared__ unsigned Alo[128 * RS];

    const int row0 = blockIdx.x * 128;

    // ---- cooperative fp32 -> packed bf16-pair hi/lo conversion into LDS ----
    for (int p = threadIdx.x; p < 128 * 64; p += 256) {
        int r = p >> 6, kp = p & 63;         // row in tile, k-pair index
        int row = row0 + r;
        unsigned hi = 0, lo = 0;
        if (row < N) {
            float2 f = *(const float2*)(X + (size_t)row * KDIM + kp * 2);
            if (relu_in) { f.x = fmaxf(f.x, 0.0f); f.y = fmaxf(f.y, 0.0f); }
            __bf16 h0 = (__bf16)f.x,              h1 = (__bf16)f.y;
            __bf16 l0 = (__bf16)(f.x - (float)h0), l1 = (__bf16)(f.y - (float)h1);
            hi = (unsigned)__builtin_bit_cast(unsigned short, h0) |
                 ((unsigned)__builtin_bit_cast(unsigned short, h1) << 16);
            lo = (unsigned)__builtin_bit_cast(unsigned short, l0) |
                 ((unsigned)__builtin_bit_cast(unsigned short, l1) << 16);
        }
        Ahi[r * RS + kp] = hi;
        Alo[r * RS + kp] = lo;
    }
    __syncthreads();

    const int lane = threadIdx.x & 31;
    const int wave = threadIdx.x >> 5;       // M-tile index within block
    const int g = lane >> 4, m = lane & 15;
    const int mrow = wave * 16 + m;          // local row owned by this lane

    if (row0 + wave * 16 >= N) return;       // wave-uniform skip (EXEC stays ~0)

    // ---- pull all A fragments (4 k-chunks, hi+lo) from LDS: 16x ds_load_b128
    v16bf ahi[KC4], alo[KC4];
#pragma unroll
    for (int kc = 0; kc < KC4; ++kc) {
        const unsigned* bh = &Ahi[mrow * RS + kc * 16 + g * 4];
        const unsigned* bl = &Alo[mrow * RS + kc * 16 + g * 4];
        v4u h0 = *(const v4u*)bh;
        v4u h1 = *(const v4u*)(bh + 8);
        v4u l0 = *(const v4u*)bl;
        v4u l1 = *(const v4u*)(bl + 8);
        v8u th, tl;
#pragma unroll
        for (int i = 0; i < 4; ++i) {
            th[i] = h0[i]; th[i + 4] = h1[i];
            tl[i] = l0[i]; tl[i + 4] = l1[i];
        }
        ahi[kc] = __builtin_bit_cast(v16bf, th);
        alo[kc] = __builtin_bit_cast(v16bf, tl);
    }

    // ---- N-tile loop: pre-packed B loads + 12 WMMAs per tile ----
#pragma unroll
    for (int nt = 0; nt < OUT / 16; ++nt) {
        v8f acc = {};
#pragma unroll
        for (int kc = 0; kc < KC4; ++kc) {
            const unsigned* p = Wp + ((((size_t)nt * KC4 + kc) * 2) * 32 + lane) * 8;
            v16bf bhi = *(const v16bf*)p;
            v16bf blo = *(const v16bf*)(p + 256);
            acc = __builtin_amdgcn_wmma_f32_16x16x32_bf16(false, ahi[kc], false, bhi,
                                                          (short)0, acc, false, false);
            acc = __builtin_amdgcn_wmma_f32_16x16x32_bf16(false, ahi[kc], false, blo,
                                                          (short)0, acc, false, false);
            acc = __builtin_amdgcn_wmma_f32_16x16x32_bf16(false, alo[kc], false, bhi,
                                                          (short)0, acc, false, false);
        }
        // D layout: lane holds N = m; VGPR r holds M = r + 8*g.
        const int col = nt * 16 + m;
        float bv = add_bias ? bias[col] : 0.0f;
#pragma unroll
        for (int r = 0; r < 8; ++r) {
            int row = row0 + wave * 16 + r + 8 * g;
            Y[(size_t)row * OUT + col] = acc[r] + bv;
        }
    }
}

// ---------------------------------------------------------------------------
// SpMM: acc[rows[e]] += support[cols[e]] * vals[e]   (one wave per edge)
// ---------------------------------------------------------------------------
template <int F>
__global__ void spmm_atomic_kernel(const int* __restrict__ rows,
                                   const int* __restrict__ cols,
                                   const float* __restrict__ vals,
                                   const float* __restrict__ support,
                                   float* __restrict__ acc, int E)
{
    int e = blockIdx.x * 8 + (threadIdx.x >> 5);
    if (e >= E) return;
    int lane = threadIdx.x & 31;
    int r = rows[e], c = cols[e];
    float w = vals[e];
    constexpr int PL = F / 32;
    const float* src = support + (size_t)c * F + lane * PL;
    float* dst = acc + (size_t)r * F + lane * PL;
    if constexpr (PL == 4) {
        float4 s = *(const float4*)src;
        atomicAdd(dst + 0, s.x * w);
        atomicAdd(dst + 1, s.y * w);
        atomicAdd(dst + 2, s.z * w);
        atomicAdd(dst + 3, s.w * w);
    } else {
        float2 s = *(const float2*)src;
        atomicAdd(dst + 0, s.x * w);
        atomicAdd(dst + 1, s.y * w);
    }
}

__global__ void relu_kernel(float* __restrict__ p, long long n)
{
    long long i = (long long)blockIdx.x * blockDim.x + threadIdx.x;
    if (i < n) p[i] = fmaxf(p[i], 0.0f);
}

// LayerNorm over feature dim (128); relu of the previous fc layer is fused
// into the load. Unbiased std, eps added to std (torch-style).
__global__ void layernorm_kernel(const float* __restrict__ in,
                                 float* __restrict__ out,
                                 const float* __restrict__ gamma,
                                 const float* __restrict__ beta, int N)
{
    int row = blockIdx.x * 8 + (threadIdx.x >> 5);
    if (row >= N) return;
    int lane = threadIdx.x & 31;
    float4 x = *(const float4*)(in + (size_t)row * 128 + lane * 4);
    x.x = fmaxf(x.x, 0.0f); x.y = fmaxf(x.y, 0.0f);
    x.z = fmaxf(x.z, 0.0f); x.w = fmaxf(x.w, 0.0f);
    float s = x.x + x.y + x.z + x.w;
    float q = x.x * x.x + x.y * x.y + x.z * x.z + x.w * x.w;
#pragma unroll
    for (int off = 16; off > 0; off >>= 1) {
        s += __shfl_xor(s, off, 32);
        q += __shfl_xor(q, off, 32);
    }
    float mean = s * (1.0f / 128.0f);
    float var  = fmaxf((q - 128.0f * mean * mean) * (1.0f / 127.0f), 0.0f);
    float inv  = 1.0f / (sqrtf(var) + 1e-6f);
    float4 gm = *(const float4*)(gamma + lane * 4);
    float4 bt = *(const float4*)(beta + lane * 4);
    float4 y;
    y.x = gm.x * (x.x - mean) * inv + bt.x;
    y.y = gm.y * (x.y - mean) * inv + bt.y;
    y.z = gm.z * (x.z - mean) * inv + bt.z;
    y.w = gm.w * (x.w - mean) * inv + bt.w;
    *(float4*)(out + (size_t)row * 128 + lane * 4) = y;
}

// ---------------------------------------------------------------------------
extern "C" void kernel_launch(void* const* d_in, const int* in_sizes, int n_in,
                              void* d_out, int out_size, void* d_ws, size_t ws_size,
                              hipStream_t stream)
{
    const float* x     = (const float*)d_in[0];
    const int*   erows = (const int*)d_in[1];
    const int*   ecols = (const int*)d_in[2];
    const float* evals = (const float*)d_in[3];
    const float* fc1_w = (const float*)d_in[4];
    const float* fc1_b = (const float*)d_in[5];
    const float* fc2_w = (const float*)d_in[6];
    const float* fc2_b = (const float*)d_in[7];
    const float* ln_g  = (const float*)d_in[8];
    const float* ln_b  = (const float*)d_in[9];
    const float* gw[4][2];
    const float* gb[4];
    for (int i = 0; i < 4; ++i) {
        gw[i][0] = (const float*)d_in[10 + i * 3];   // wn
        gw[i][1] = (const float*)d_in[11 + i * 3];   // ws
        gb[i]    = (const float*)d_in[12 + i * 3];   // b
    }
    const int N = in_sizes[0] / 128;   // 100000 (multiple of 16)
    const int E = in_sizes[1];         // 3200000

    // -------- workspace layout --------
    char* ws = (char*)d_ws;
    size_t bufBytes = (size_t)N * 128 * sizeof(float);     // 51.2 MB
    float* bufA = (float*)ws;
    float* bufB = (float*)(ws + bufBytes);
    float* bufC = (float*)(ws + 2 * bufBytes);
    unsigned* wpack = (unsigned*)(ws + 3 * bufBytes);
    const size_t sz128 = (size_t)8 * KC4 * 2 * 32 * 8;     // dwords per 128-out weight
    const size_t sz64  = (size_t)4 * KC4 * 2 * 32 * 8;
    unsigned* p_fc1 = wpack;
    unsigned* p_fc2 = p_fc1 + sz128;
    unsigned* p_g[4][2];
    unsigned* cur = p_fc2 + sz128;
    for (int i = 0; i < 4; ++i)
        for (int j = 0; j < 2; ++j) { p_g[i][j] = cur; cur += (i == 3 ? sz64 : sz128); }

    // -------- pack all weights into WMMA fragment layout (hi/lo bf16) --------
    auto pack = [&](const float* W, unsigned* dst, int OUT, int tr) {
        int total = (OUT / 16) * KC4 * 32 * 8;
        pack_weight_kernel<<<(total + 255) / 256, 256, 0, stream>>>(W, dst, OUT, tr);
    };
    pack(fc1_w, p_fc1, 128, 1);
    pack(fc2_w, p_fc2, 128, 1);
    for (int i = 0; i < 4; ++i) {
        int OUT = (i == 3) ? 64 : 128;
        pack(gw[i][0], p_g[i][0], OUT, 0);
        pack(gw[i][1], p_g[i][1], OUT, 0);
    }

    const dim3 g128((N + 127) / 128);
    const int spmmBlocks = (E + 7) / 8;

    // fc1 -> bufA ; fc2 (relu_in fused) -> bufB ; LN (relu fused) -> bufA
    gemm_wmma_kernel<128><<<g128, 256, 0, stream>>>(x,    p_fc1, fc1_b, bufA, N, 1, 0);
    gemm_wmma_kernel<128><<<g128, 256, 0, stream>>>(bufA, p_fc2, fc2_b, bufB, N, 1, 1);
    layernorm_kernel<<<(N + 7) / 8, 256, 0, stream>>>(bufB, bufA, ln_g, ln_b, N);

    // gc1..gc3 (128-wide); relu of previous layer fused into gemm input
    float* h = bufA; float* S = bufB; float* Acc = bufC;
    for (int i = 0; i < 3; ++i) {
        int rin = (i > 0) ? 1 : 0;   // gc1 input is LN output (no relu)
        gemm_wmma_kernel<128><<<g128, 256, 0, stream>>>(h, p_g[i][0], nullptr, S,   N, 0, rin);
        gemm_wmma_kernel<128><<<g128, 256, 0, stream>>>(h, p_g[i][1], gb[i],   Acc, N, 1, rin);
        spmm_atomic_kernel<128><<<spmmBlocks, 256, 0, stream>>>(erows, ecols, evals, S, Acc, E);
        float* oldh = h; h = Acc; Acc = S; S = oldh;
    }

    // gc4 (64-wide) -> d_out; final relu applied explicitly
    float* out = (float*)d_out;
    gemm_wmma_kernel<64><<<g128, 256, 0, stream>>>(h, p_g[3][0], nullptr, S,   N, 0, 1);
    gemm_wmma_kernel<64><<<g128, 256, 0, stream>>>(h, p_g[3][1], gb[3],   out, N, 1, 1);
    spmm_atomic_kernel<64><<<spmmBlocks, 256, 0, stream>>>(erows, ecols, evals, S, out, E);
    long long nOut = (long long)N * 64;
    relu_kernel<<<(int)((nOut + 255) / 256), 256, 0, stream>>>(out, nOut);
}
